// AdvancedHetDDIv2_21620865368736
// MI455X (gfx1250) — matrix operand order repaired and checked by reading, not compiled, not run on hardware
//
#include <hip/hip_runtime.h>
#include <hip/hip_bf16.h>

typedef __attribute__((ext_vector_type(16))) _Float16 v16h;
typedef __attribute__((ext_vector_type(8)))  _Float16 v8h;
typedef __attribute__((ext_vector_type(4)))  _Float16 v4h;
typedef __attribute__((ext_vector_type(8)))  float    v8f;

#define HD   256   // hidden
#define NHD  4     // heads
#define DHD  64    // head dim
#define AT   64    // atoms per molecule
#define EPM  512   // edges per molecule

// 2-bit block bit-reverse: 0->0, 1->2, 2->1, 3->3.  Within a 32-col group the
// f16 A-fragment K order is {0..7,16..23 | 8..15,24..31}; storing column c at
// position (c&7)|(permblk(c>>3)<<3) makes each fragment 16 contiguous halves.
__device__ __forceinline__ int permblk(int b) { return ((b & 1) << 1) | (b >> 1); }

// ---------------------------------------------------------------------------
// Tiled GEMM: C[M,N] = A[M,K] @ B[K,N] (+ bias[N]), fp32 io, f16 WMMA fp32-acc.
// Requires M%64==0, N%32==0, K%32==0 (all call sites satisfy this).
// Block tile 64x32, 8 waves (4x2), one 16x16 WMMA tile per wave, K-step 32.
// A staged unpermuted (contiguous k), B staged transposed: all operand
// fetches are single aligned v16h LDS loads (2x ds_load_b128).
// ---------------------------------------------------------------------------
template <bool GATHER>
__global__ __launch_bounds__(256) void gemm_wmma(
    const float* __restrict__ A, const float* __restrict__ B,
    const float* __restrict__ bias, float* __restrict__ C,
    const int* __restrict__ arows,
    int M, int N, int K, long strideB, long strideC)
{
  __shared__ alignas(32) _Float16 sA2[64][48];  // fragment-order, stride 96B (32B-mult)
  __shared__ alignas(32) _Float16 sBt[32][48];  // transposed: [n][k]
  const float* Bp = B + (size_t)blockIdx.z * strideB;
  float*       Cp = C + (size_t)blockIdx.z * strideC;
  int t    = threadIdx.x;
  int lane = t & 31, wave = t >> 5;
  int wm   = wave >> 1, wn = wave & 1;
  int half = lane >> 4, l16 = lane & 15;
  int bM   = blockIdx.y * 64, bN = blockIdx.x * 32;

  // A staging geometry: thread owns 8 consecutive cols of one tile row
  int arow = t >> 2;                 // 0..63
  int ablk = t & 3;                  // 8-col block
  int pbase = ablk * 8;              // LDS position of that block (contiguous k)
  int grow = bM + arow;
  const float* baseA = A + (size_t)(GATHER ? arows[grow] : grow) * K + ablk * 8;
  // B staging geometry: thread owns 4 consecutive cols of one tile row
  int brow = t >> 3;                 // k within tile, 0..31
  int bcol = (t & 7) * 4;
  const float* baseB = Bp + bN + bcol;

  v8f acc = {0.f,0.f,0.f,0.f,0.f,0.f,0.f,0.f};

  for (int k0 = 0; k0 < K; k0 += 32) {
    __syncthreads();
    {
      const float4* pa = (const float4*)(baseA + k0);
      float4 f0 = pa[0], f1 = pa[1];
      v8h h8;
      h8[0]=(_Float16)f0.x; h8[1]=(_Float16)f0.y; h8[2]=(_Float16)f0.z; h8[3]=(_Float16)f0.w;
      h8[4]=(_Float16)f1.x; h8[5]=(_Float16)f1.y; h8[6]=(_Float16)f1.z; h8[7]=(_Float16)f1.w;
      *(v8h*)&sA2[arow][pbase] = h8;                 // one ds_store_b128
    }
    {
      float4 g = *(const float4*)(baseB + (size_t)(k0 + brow) * N);
      sBt[bcol + 0][brow] = (_Float16)g.x;
      sBt[bcol + 1][brow] = (_Float16)g.y;
      sBt[bcol + 2][brow] = (_Float16)g.z;
      sBt[bcol + 3][brow] = (_Float16)g.w;
    }
    __syncthreads();

    v16h av = *(const v16h*)&sA2[wm * 16 + l16][half * 16];
    v16h bw = *(const v16h*)&sBt[wn * 16 + l16][half * 16];
    acc = __builtin_amdgcn_wmma_f32_16x16x32_f16(false, av, false, bw,
                                                 (short)0, acc, false, false);
  }

  int n = bN + wn * 16 + l16;
  float bb = bias ? bias[n] : 0.f;
  #pragma unroll
  for (int r = 0; r < 8; ++r) {
    int m = bM + wm * 16 + half * 8 + r;
    Cp[(size_t)m * N + n] = acc[r] + bb;
  }
}

// ---------------------------------------------------------------------------
// GATv2 edge stage, one block per molecule; fs/fd in LDS, fd recycled as the
// aggregation accumulator (ds_add atomics). h_out = agg + gb, in place of h.
// ---------------------------------------------------------------------------
__global__ __launch_bounds__(256) void gat_edge(
    const float* __restrict__ fs, const float* __restrict__ fd,
    const int* __restrict__ esrc, const int* __restrict__ edst,
    const float* __restrict__ ga, const float* __restrict__ gb,
    float* __restrict__ hout)
{
  __shared__ alignas(16) float sfs[AT][HD];      // 64 KB
  __shared__ alignas(16) float sfd[AT][HD];      // 64 KB, reused as agg
  __shared__ int   ssrc[EPM];
  __shared__ int   sdst[EPM];
  __shared__ float slog[EPM][NHD];               // 8 KB
  __shared__ float sga[HD];
  __shared__ float sgb[HD];

  int mol = blockIdx.x;
  int t   = threadIdx.x;
  const float4* fsp4 = (const float4*)(fs + (size_t)mol * AT * HD);
  const float4* fdp4 = (const float4*)(fd + (size_t)mol * AT * HD);

  #pragma unroll
  for (int i = 0; i < 16; ++i) {
    int id4 = t + i * 256;                       // 4096 float4 per tile
    ((float4*)sfs)[id4] = fsp4[id4];
    ((float4*)sfd)[id4] = fdp4[id4];
  }
  sga[t] = ga[t];
  sgb[t] = gb[t];
  #pragma unroll
  for (int i = 0; i < 2; ++i) {
    int e = t + i * 256;
    ssrc[e] = esrc[mol * EPM + e] & (AT - 1);
    sdst[e] = edst[mol * EPM + e] & (AT - 1);
  }
  __syncthreads();

  // Phase A: per-(edge,head) logits (2048 tasks, 8/thread), float4 inner dot
  #pragma unroll
  for (int i = 0; i < 8; ++i) {
    int id = t + i * 256;
    int e = id >> 2, hh = id & 3;
    const float4* ps = (const float4*)(sfs[ssrc[e]] + hh * DHD);
    const float4* pd = (const float4*)(sfd[sdst[e]] + hh * DHD);
    const float4* pg = (const float4*)(sga + hh * DHD);
    float sum = 0.f;
    #pragma unroll
    for (int d4 = 0; d4 < 16; ++d4) {
      float4 a = ps[d4], b = pd[d4], g = pg[d4];
      float x;
      x = a.x + b.x; x = (x > 0.f) ? x : 0.2f * x; sum += x * g.x;
      x = a.y + b.y; x = (x > 0.f) ? x : 0.2f * x; sum += x * g.y;
      x = a.z + b.z; x = (x > 0.f) ? x : 0.2f * x; sum += x * g.z;
      x = a.w + b.w; x = (x > 0.f) ? x : 0.2f * x; sum += x * g.w;
    }
    slog[e][hh] = sum;
  }
  __syncthreads();

  // zero agg (reuses sfd; Phase A was its last reader)
  #pragma unroll
  for (int i = 0; i < 64; ++i) ((float*)sfd)[t + i * 256] = 0.f;

  // Phase B: softmax over incoming edges; slot = (atom,head) per thread
  {
    int a = t >> 2, hh = t & 3;
    float mx = -1e30f;
    for (int e = 0; e < EPM; ++e)
      if (sdst[e] == a) mx = fmaxf(mx, slog[e][hh]);
    float sm = 0.f;
    for (int e = 0; e < EPM; ++e)
      if (sdst[e] == a) { float ex = __expf(slog[e][hh] - mx); slog[e][hh] = ex; sm += ex; }
    float inv = (sm > 0.f) ? 1.f / sm : 0.f;
    for (int e = 0; e < EPM; ++e)
      if (sdst[e] == a) slog[e][hh] *= inv;     // unique owner per (e,hh)
  }
  __syncthreads();

  // Phase C: aggregation; thread owns dim t, loop edges, LDS atomics
  {
    int hh = t >> 6;
    for (int e = 0; e < EPM; ++e)
      atomicAdd(&sfd[sdst[e]][t], slog[e][hh] * sfs[ssrc[e]][t]);
  }
  __syncthreads();

  float4* out4 = (float4*)(hout + (size_t)mol * AT * HD);
  #pragma unroll
  for (int i = 0; i < 16; ++i) {
    int id4 = t + i * 256;
    int c4 = (id4 & 63) * 4;
    float4 vgl = ((float4*)sfd)[id4];
    vgl.x += sgb[c4 + 0]; vgl.y += sgb[c4 + 1];
    vgl.z += sgb[c4 + 2]; vgl.w += sgb[c4 + 3];
    out4[id4] = vgl;
  }
}

// ---------------------------------------------------------------------------
// Cross-modal attention, one block per drug. q staged row-major (A-fragment
// order), k row-major (B-fragments contiguous in K), v transposed [dim][key].
// 16 strips = (head, m-tile); each wave: S=QK^T (WMMA), row softmax, P via
// LDS (row-major), ctx = P*V (WMMA).
// ---------------------------------------------------------------------------
__global__ __launch_bounds__(256) void cross_attn(
    const float* __restrict__ q, const float* __restrict__ k,
    const float* __restrict__ v, float* __restrict__ ctx)
{
  __shared__ alignas(32) _Float16 sqp[AT][HD + 16];   // row-major, stride 544B
  __shared__ alignas(32) _Float16 sk [AT][HD + 16];   // row-major
  __shared__ alignas(32) _Float16 svT[HD][AT + 16];   // [dim][key], stride 160B
  __shared__ alignas(32) _Float16 sp [8][16][AT + 16];

  int d    = blockIdx.x;
  int t    = threadIdx.x;
  int lane = t & 31, wave = t >> 5;
  int half = lane >> 4, l16 = lane & 15;
  size_t base = (size_t)d * AT * HD;

  #pragma unroll
  for (int i = 0; i < 16; ++i) {
    int id4 = t + i * 256;                 // 4096 float4
    int row = id4 >> 6;                    // 64 float4 per row
    int c4  = (id4 & 63) * 4;
    float4 fq = *(const float4*)(q + base + (size_t)row * HD + c4);
    float4 fk = *(const float4*)(k + base + (size_t)row * HD + c4);
    float4 fv = *(const float4*)(v + base + (size_t)row * HD + c4);
    int kb = c4 & 31, grp = c4 >> 5;
    int p  = kb;
    v4h hq; hq[0]=(_Float16)fq.x; hq[1]=(_Float16)fq.y; hq[2]=(_Float16)fq.z; hq[3]=(_Float16)fq.w;
    *(v4h*)&sqp[row][grp * 32 + p] = hq;   // ds_store_b64
    v4h hk; hk[0]=(_Float16)fk.x; hk[1]=(_Float16)fk.y; hk[2]=(_Float16)fk.z; hk[3]=(_Float16)fk.w;
    *(v4h*)&sk[row][c4] = hk;              // ds_store_b64
    svT[c4 + 0][row] = (_Float16)fv.x;
    svT[c4 + 1][row] = (_Float16)fv.y;
    svT[c4 + 2][row] = (_Float16)fv.z;
    svT[c4 + 3][row] = (_Float16)fv.w;
  }
  __syncthreads();

  for (int s8 = 0; s8 < 2; ++s8) {
    int strip = wave + s8 * 8;      // 0..15
    int h  = strip >> 2;            // head
    int mt = strip & 3;             // query m-tile
    v8f zero = {0.f,0.f,0.f,0.f,0.f,0.f,0.f,0.f};
    v8f acc[4] = {zero, zero, zero, zero};

    // S = Q @ K^T (16-row strip x 64 keys)
    #pragma unroll
    for (int kt = 0; kt < 2; ++kt) {
      int kk0 = h * DHD + kt * 32;
      v16h av = *(const v16h*)&sqp[mt * 16 + l16][kk0 + half * 16];
      #pragma unroll
      for (int nt = 0; nt < 4; ++nt) {
        v16h bw = *(const v16h*)&sk[nt * 16 + l16][kk0 + half * 16];
        acc[nt] = __builtin_amdgcn_wmma_f32_16x16x32_f16(false, av, false, bw,
                                                         (short)0, acc[nt], false, false);
      }
    }

    // scale + row softmax (rows live in 16-lane groups; xor-reduce <=8)
    #pragma unroll
    for (int r = 0; r < 8; ++r) {
      float mx = -1e30f;
      #pragma unroll
      for (int nt = 0; nt < 4; ++nt) { acc[nt][r] *= 0.125f; mx = fmaxf(mx, acc[nt][r]); }
      #pragma unroll
      for (int m = 8; m >= 1; m >>= 1) mx = fmaxf(mx, __shfl_xor(mx, m, 32));
      float sm = 0.f;
      #pragma unroll
      for (int nt = 0; nt < 4; ++nt) { float ex = __expf(acc[nt][r] - mx); acc[nt][r] = ex; sm += ex; }
      #pragma unroll
      for (int m = 8; m >= 1; m >>= 1) sm += __shfl_xor(sm, m, 32);
      float inv = 1.f / sm;
      int ml = half * 8 + r;
      #pragma unroll
      for (int nt = 0; nt < 4; ++nt) {
        float pv = acc[nt][r] * inv;
        int col = nt * 16 + l16;
        int kb = col & 31, grp = col >> 5;
        int p  = kb;
        sp[wave][ml][grp * 32 + p] = (_Float16)pv;   // contiguous k for A-fragments
      }
    }

    // ctx strip = P @ V_h (same wave wrote sp[wave]; dscnt waits cover RAW)
    v8f acc2[4] = {zero, zero, zero, zero};
    #pragma unroll
    for (int kt = 0; kt < 2; ++kt) {
      int kk0 = kt * 32;
      v16h av = *(const v16h*)&sp[wave][l16][kk0 + half * 16];
      #pragma unroll
      for (int nt = 0; nt < 4; ++nt) {
        v16h bw = *(const v16h*)&svT[h * DHD + nt * 16 + l16][kk0 + half * 16];
        acc2[nt] = __builtin_amdgcn_wmma_f32_16x16x32_f16(false, av, false, bw,
                                                          (short)0, acc2[nt], false, false);
      }
    }
    #pragma unroll
    for (int r = 0; r < 8; ++r) {
      int m = mt * 16 + half * 8 + r;
      #pragma unroll
      for (int nt = 0; nt < 4; ++nt)
        ctx[base + (size_t)m * HD + h * DHD + nt * 16 + l16] = acc2[nt][r];
    }
    __syncthreads();   // uniform: every wave runs exactly 2 strips
  }
}

// ---------------------------------------------------------------------------
// drug[d] = mean over atoms of (h + attn)
// ---------------------------------------------------------------------------
__global__ __launch_bounds__(256) void drug_mean(
    const float* __restrict__ h, const float* __restrict__ attn,
    float* __restrict__ drug)
{
  int d = blockIdx.x, c = threadIdx.x;
  float s = 0.f;
  for (int a = 0; a < AT; ++a) {
    size_t idx = ((size_t)d * AT + a) * HD + c;
    s += h[idx] + attn[idx];
  }
  drug[(size_t)d * HD + c] = s * (1.f / 64.f);
}

// ---------------------------------------------------------------------------
// Neural tensor interaction: one wave per pair, float4 throughout.
// ts[p,k] = dot(TD[k][left], drug[right]); lin from Vw; tanh; dot with ow.
// ---------------------------------------------------------------------------
__global__ __launch_bounds__(256) void nti_kernel(
    const float* __restrict__ TD, const float* __restrict__ drug,
    const int* __restrict__ left, const int* __restrict__ right,
    const float* __restrict__ Vw, const float* __restrict__ Vb,
    const float* __restrict__ ow, const float* __restrict__ ob,
    float* __restrict__ out)
{
  int wave = threadIdx.x >> 5, lane = threadIdx.x & 31;
  int p = blockIdx.x * 8 + wave;
  int l = left[p], r = right[p];
  const float* Ae = drug + (size_t)l * HD;
  const float* Be = drug + (size_t)r * HD;

  float4 a0 = *(const float4*)(Ae + lane * 8);
  float4 a1 = *(const float4*)(Ae + lane * 8 + 4);
  float4 b0 = *(const float4*)(Be + lane * 8);
  float4 b1 = *(const float4*)(Be + lane * 8 + 4);

  float tot[16];
  #pragma unroll
  for (int kk = 0; kk < 16; ++kk) {
    const float4* td = (const float4*)(TD + ((size_t)kk * 512 + l) * HD + lane * 8);
    float4 t0 = td[0], t1 = td[1];
    tot[kk] = t0.x*b0.x + t0.y*b0.y + t0.z*b0.z + t0.w*b0.w
            + t1.x*b1.x + t1.y*b1.y + t1.z*b1.z + t1.w*b1.w;
  }
  const float av8[8] = {a0.x,a0.y,a0.z,a0.w,a1.x,a1.y,a1.z,a1.w};
  const float bv8[8] = {b0.x,b0.y,b0.z,b0.w,b1.x,b1.y,b1.z,b1.w};
  #pragma unroll
  for (int j = 0; j < 8; ++j) {
    int g = lane * 8 + j;
    const float4* vwA = (const float4*)(Vw + (size_t)g * 16);
    const float4* vwB = (const float4*)(Vw + (size_t)(256 + g) * 16);
    #pragma unroll
    for (int kq = 0; kq < 4; ++kq) {
      float4 wa = vwA[kq], wb = vwB[kq];
      tot[kq*4+0] += av8[j]*wa.x + bv8[j]*wb.x;
      tot[kq*4+1] += av8[j]*wa.y + bv8[j]*wb.y;
      tot[kq*4+2] += av8[j]*wa.z + bv8[j]*wb.z;
      tot[kq*4+3] += av8[j]*wa.w + bv8[j]*wb.w;
    }
  }
  #pragma unroll
  for (int kk = 0; kk < 16; ++kk)
    #pragma unroll
    for (int m = 16; m >= 1; m >>= 1) tot[kk] += __shfl_xor(tot[kk], m, 32);

  float o = ob[0];
  #pragma unroll
  for (int kk = 0; kk < 16; ++kk) o += tanhf(tot[kk] + Vb[kk]) * ow[kk];
  if (lane == 0) out[p] = o;
}

// ---------------------------------------------------------------------------
extern "C" void kernel_launch(void* const* d_in, const int* in_sizes, int n_in,
                              void* d_out, int out_size, void* d_ws, size_t ws_size,
                              hipStream_t stream) {
  (void)in_sizes; (void)n_in; (void)out_size; (void)ws_size;
  const float* atom_feat     = (const float*)d_in[0];
  const int*   edge_src      = (const int*)  d_in[1];
  const int*   edge_dst      = (const int*)  d_in[2];
  const float* kg_node_feats = (const float*)d_in[3];
  /* d_in[4] = edge_type, unused per reference */
  const int*   kg_neighbors  = (const int*)  d_in[5];
  const int*   left          = (const int*)  d_in[6];
  const int*   right         = (const int*)  d_in[7];
  const float* Wi  = (const float*)d_in[8];
  const float* bi  = (const float*)d_in[9];
  const float* gWl = (const float*)d_in[10];
  const float* gWr = (const float*)d_in[11];
  const float* ga  = (const float*)d_in[12];
  const float* gb  = (const float*)d_in[13];
  const float* Wkp = (const float*)d_in[14];
  const float* bkp = (const float*)d_in[15];
  const float* Wq  = (const float*)d_in[16];
  const float* bq  = (const float*)d_in[17];
  const float* Wk  = (const float*)d_in[18];
  const float* bk  = (const float*)d_in[19];
  const float* Wv  = (const float*)d_in[20];
  const float* bv  = (const float*)d_in[21];
  const float* Wo  = (const float*)d_in[22];
  const float* bo  = (const float*)d_in[23];
  const float* ntW = (const float*)d_in[24];
  const float* Vw  = (const float*)d_in[25];
  const float* Vb  = (const float*)d_in[26];
  const float* ow  = (const float*)d_in[27];
  const float* ob  = (const float*)d_in[28];
  float* out = (float*)d_out;

  const int Natoms = 32768, D = 512, P = 4096;
  const size_t SZ = (size_t)Natoms * HD;
  float* ws    = (float*)d_ws;
  float* bufH  = ws;            // h (updated in place by GAT)
  float* bufFS = ws + 1 * SZ;   // fs / kgp / ctx
  float* bufFD = ws + 2 * SZ;   // fd / q / attn
  float* buf3  = ws + 3 * SZ;   // k  / TD
  float* buf4  = ws + 4 * SZ;   // v
  float* drug  = ws + 5 * SZ;   // 512*256

  dim3 blk(256);
  auto gemm = [&](const float* A, const float* B, const float* bias, float* C,
                  int M, int N, int K, int batch, long sB, long sC) {
    dim3 grid((N + 31) / 32, (M + 63) / 64, batch);
    gemm_wmma<false><<<grid, blk, 0, stream>>>(A, B, bias, C, nullptr, M, N, K, sB, sC);
  };

  // 1. input projection: h = atom_feat @ Wi + bi
  gemm(atom_feat, Wi, bi, bufH, Natoms, HD, 128, 1, 0, 0);

  // 2. three GATv2 layers
  for (int l = 0; l < 3; ++l) {
    gemm(bufH, gWl + (size_t)l * HD * HD, nullptr, bufFS, Natoms, HD, HD, 1, 0, 0);
    gemm(bufH, gWr + (size_t)l * HD * HD, nullptr, bufFD, Natoms, HD, HD, 1, 0, 0);
    gat_edge<<<dim3(D), blk, 0, stream>>>(bufFS, bufFD, edge_src, edge_dst,
                                          ga + l * NHD * DHD, gb + l * HD, bufH);
  }

  // 3. KG projection, only the 32768 gathered rows (row gather = kg_neighbors)
  {
    dim3 grid((HD + 31) / 32, (Natoms + 63) / 64, 1);
    gemm_wmma<true><<<grid, blk, 0, stream>>>(kg_node_feats, Wkp, bkp, bufFS,
                                              kg_neighbors, Natoms, HD, HD, 0, 0);
  }

  // 4-6. q from atoms, k/v from gathered kg rows
  gemm(bufH,  Wq, bq, bufFD, Natoms, HD, HD, 1, 0, 0);
  gemm(bufFS, Wk, bk, buf3,  Natoms, HD, HD, 1, 0, 0);
  gemm(bufFS, Wv, bv, buf4,  Natoms, HD, HD, 1, 0, 0);

  // 7. per-drug cross attention -> ctx (reuse bufFS)
  cross_attn<<<dim3(D), blk, 0, stream>>>(bufFD, buf3, buf4, bufFS);

  // 8. output projection: attn = ctx @ Wo + bo (reuse bufFD)
  gemm(bufFS, Wo, bo, bufFD, Natoms, HD, HD, 1, 0, 0);

  // 9. drug embeddings = mean(atoms + attn)
  drug_mean<<<dim3(D), blk, 0, stream>>>(bufH, bufFD, drug);

  // 10. TD[k] = drug @ ntW[k], batched over 16 slices (reuse buf3)
  gemm(drug, ntW, nullptr, buf3, D, HD, HD, 16, (long)HD * HD, (long)D * HD);

  // 11. pairwise neural tensor interaction
  nti_kernel<<<dim3(P / 8), blk, 0, stream>>>(buf3, drug, left, right,
                                              Vw, Vb, ow, ob, out);
}
